// Model_bs_LSTM_37563783971049
// MI455X (gfx1250) — matrix operand order repaired and checked by reading, not compiled, not run on hardware
//
#include <hip/hip_runtime.h>

typedef __attribute__((ext_vector_type(16))) _Float16     v16h;
typedef __attribute__((ext_vector_type(8)))  _Float16     v8h;
typedef __attribute__((ext_vector_type(8)))  float        v8f;
typedef __attribute__((ext_vector_type(4)))  unsigned int v4u;
typedef __attribute__((ext_vector_type(8)))  int          v8i;
typedef __attribute__((ext_vector_type(4)))  int          v4i;

namespace {

constexpr int B_   = 64;
constexpr int S_   = 2048;
constexpr int C_   = 32;
constexpr int WIN_ = 20;
constexpr int T_   = S_ - WIN_;    // 2028 timesteps
constexpr int T2_  = 2032;         // padded to multiple of 16
constexpr int H_   = 128;
constexpr int G_   = 512;          // 4*H (i,f,g,o)
constexpr int DP_  = 320;          // projection dim
constexpr int DIN_ = 640;          // WIN*C
constexpr int LBL_ = 48;
constexpr long M2_ = (long)B_ * T2_;  // 130048 padded rows

__device__ inline v16h cvt16(v8f lo, v8f hi) {
  v16h f;
#pragma unroll
  for (int j = 0; j < 8; ++j) { f[j] = (_Float16)lo[j]; f[j + 8] = (_Float16)hi[j]; }
  return f;
}
__device__ inline v16h cat16(v8h lo, v8h hi) {
  v16h f;
#pragma unroll
  for (int j = 0; j < 8; ++j) { f[j] = lo[j]; f[j + 8] = hi[j]; }
  return f;
}

// A-fragment (16x32 f16), ISA 7.12.2: elems 0..7 -> K=hi+0..7, 8..15 -> K=hi+16..23
__device__ inline v16h load_a_frag(const float* q, int lane) {
  const int hi = ((lane >> 4) & 1) << 3;
  return cvt16(*(const v8f*)(q + hi), *(const v8f*)(q + 16 + hi));
}
__device__ inline v16h load_a_frag(const _Float16* q, int lane) {
  const int hi = ((lane >> 4) & 1) << 3;
  return cat16(*(const v8h*)(q + hi), *(const v8h*)(q + 16 + hi));
}
// B-fragment (32x16 f16) from row-major W[N][K]: elems 0..7 -> K=hi+0..7, 8..15 -> K=hi+16..23
__device__ inline v16h load_b_frag(const _Float16* q, int lane) {
  const int hi = ((lane >> 4) & 1) << 3;
  return cat16(*(const v8h*)(q + hi), *(const v8h*)(q + 16 + hi));
}

// ---------------------------------------------------------------------------
// f32 -> f16 weight conversion (run once per launch; weights are ~1.65 MB).
// ---------------------------------------------------------------------------
__global__ __launch_bounds__(256) void k_cvt(
    const float* __restrict__ s, _Float16* __restrict__ d, int n)
{
  const int i = blockIdx.x * 256 + threadIdx.x;
  if (i < n) d[i] = (_Float16)s[i];
}

// ---------------------------------------------------------------------------
// WMMA GEMM: out[M,N] = A[M,K] * W16[N,K]^T + bias0 + bias1
// One wave per (16 x NBLK*16) strip: A-fragment reused for NBLK WMMAs.
// ---------------------------------------------------------------------------
template <typename AT, typename OT, int NBLK>
__global__ __launch_bounds__(256) void k_gemm(
    const AT* __restrict__ A, long lda,
    const _Float16* __restrict__ W, long ldw,
    const float* __restrict__ b0, const float* __restrict__ b1,
    OT* __restrict__ out, long ldo,
    int Mtiles, int Nstrips, int K)
{
  const int wave = blockIdx.x * 8 + (threadIdx.x >> 5);
  if (wave >= Mtiles * Nstrips) return;          // wave-uniform: EXEC stays full
  const int tm = wave / Nstrips, ts = wave % Nstrips;
  const int lane = threadIdx.x & 31;
  const AT* arow = A + (size_t)(tm * 16 + (lane & 15)) * lda;
  const _Float16* wrow = W + (size_t)(ts * NBLK * 16 + (lane & 15)) * ldw;
  v8f acc[NBLK];
#pragma unroll
  for (int i = 0; i < NBLK; ++i) acc[i] = (v8f){};
  for (int k = 0; k < K; k += 32) {
    v16h a = load_a_frag(arow + k, lane);
#pragma unroll
    for (int i = 0; i < NBLK; ++i) {
      v16h b = load_b_frag(wrow + (size_t)i * 16 * ldw + k, lane);
      acc[i] = __builtin_amdgcn_wmma_f32_16x16x32_f16(false, a, false, b, (short)0, acc[i], false, false);
    }
  }
#pragma unroll
  for (int i = 0; i < NBLK; ++i) {
    const int n = (ts * NBLK + i) * 16 + (lane & 15);
    const float bs = (b0 ? b0[n] : 0.f) + (b1 ? b1[n] : 0.f);
#pragma unroll
    for (int r = 0; r < 8; ++r) {
      const int m = tm * 16 + r + ((lane >> 4) << 3);
      out[(size_t)m * ldo + n] = (OT)(acc[i][r] + bs);
    }
  }
}

// ---------------------------------------------------------------------------
// Fused unfold + projection (window K-chunk kw == inputs[b, t+kw, 0:32]).
// NBLK=2 n-tiles per wave; output stored f16.
// ---------------------------------------------------------------------------
__global__ __launch_bounds__(256) void k_proj(
    const float* __restrict__ inp,      // [B,S,32] f32
    const _Float16* __restrict__ W,     // [320,640] f16 (preconverted)
    const float* __restrict__ bias,     // [320]
    _Float16* __restrict__ out)         // [M2,320] f16
{
  const int Nstrips = DP_ / 32;                  // 10 strips of 2 n-tiles
  const int wave = blockIdx.x * 8 + (threadIdx.x >> 5);
  if (wave >= (int)(M2_ / 16) * Nstrips) return;
  const int tm = wave / Nstrips, ts = wave % Nstrips;
  const int lane = threadIdx.x & 31;
  const int m0 = tm * 16;
  const int b  = m0 / T2_;                       // tile never straddles b
  int t = (m0 % T2_) + (lane & 15);
  if (t > T_ - 1) t = T_ - 1;                    // clamp padded rows (loads only)
  const _Float16* wrow = W + (size_t)(ts * 32 + (lane & 15)) * DIN_;
  v8f acc0 = {}, acc1 = {};
#pragma unroll 4
  for (int kw = 0; kw < WIN_; ++kw) {
    v16h a  = load_a_frag(inp + ((size_t)b * S_ + t + kw) * C_, lane);
    v16h w0 = load_b_frag(wrow + kw * 32, lane);
    v16h w1 = load_b_frag(wrow + (size_t)16 * DIN_ + kw * 32, lane);
    acc0 = __builtin_amdgcn_wmma_f32_16x16x32_f16(false, a, false, w0, (short)0, acc0, false, false);
    acc1 = __builtin_amdgcn_wmma_f32_16x16x32_f16(false, a, false, w1, (short)0, acc1, false, false);
  }
#pragma unroll
  for (int i = 0; i < 2; ++i) {
    const int n = ts * 32 + i * 16 + (lane & 15);
    const float bs = bias[n];
    const v8f& a = i ? acc1 : acc0;
#pragma unroll
    for (int r = 0; r < 8; ++r) {
      const int m = m0 + r + ((lane >> 4) << 3);
      out[(size_t)m * DP_ + n] = (_Float16)(a[r] + bs);
    }
  }
}

// ---------------------------------------------------------------------------
// TDM staging of one [16 x 512] f16 gate tile into LDS (double-buffered).
// Descriptor per ISA ch.8: 2D tile, data_size=1 (2B), dim0 stride = T2*512.
// This toolchain exposes the 6-argument builtin form (extra groups zeroed).
// ---------------------------------------------------------------------------
#if defined(__gfx1250__) && __has_builtin(__builtin_amdgcn_tensor_load_to_lds)
#define USE_TDM 1
#else
#define USE_TDM 0
#endif

__device__ inline void stage_gpre(_Float16* dst_lds, const _Float16* src, int tid) {
#if USE_TDM
  if ((tid >> 5) == 0) {                         // one wave issues the DMA
    const unsigned long long ga = (unsigned long long)(size_t)src;
    const unsigned long long stride = (unsigned long long)T2_ * G_;  // elements
    v4u g0;
    g0[0] = 1u;                                          // count=1 (valid user D#)
    g0[1] = (unsigned)(size_t)dst_lds;                   // lds_addr (bytes)
    g0[2] = (unsigned)(ga & 0xffffffffu);                // global_addr[31:0]
    g0[3] = (unsigned)((ga >> 32) & 0x1ffffffu) | (2u << 30);  // addr[56:32], type=2
    v8i g1;
    g1[0] = (int)(1u << 16);                             // data_size=1 (2B), mask=0
    g1[1] = (int)((unsigned)G_ << 16);                   // tensor_dim0[15:0]=512
    g1[2] = (int)(16u << 16);                            // tensor_dim1[15:0]=16
    g1[3] = (int)((unsigned)G_ << 16);                   // tile_dim0=512
    g1[4] = 16;                                          // tile_dim1=16, tile_dim2=0
    g1[5] = (int)(stride & 0xffffffffu);                 // tensor_dim0_stride lo
    g1[6] = (int)((stride >> 32) & 0xffffu);             // tensor_dim0_stride hi
    g1[7] = 0;                                           // tensor_dim1_stride (unused)
    v4i z4 = {0, 0, 0, 0};
    v8i z8 = {0, 0, 0, 0, 0, 0, 0, 0};
    __builtin_amdgcn_tensor_load_to_lds(g0, g1, z4, z4, z8, 0);
  }
#else
  const int row = tid >> 4, c0 = (tid & 15) * 32;        // 32 f16 per thread
  const v8h* s = (const v8h*)(src + (size_t)row * ((size_t)T2_ * G_) + c0);
  v8h* d = (v8h*)(dst_lds + row * G_ + c0);
#pragma unroll
  for (int u = 0; u < 4; ++u) d[u] = s[u];
#endif
}
__device__ inline void stage_wait(int tid) {
#if USE_TDM
  if ((tid >> 5) == 0) __builtin_amdgcn_s_wait_tensorcnt(0);
#endif
}

// ---------------------------------------------------------------------------
// LSTM recurrence: 4 persistent workgroups x 16 batch rows. w_hh lives in
// VGPRs (16 persistent f16 B-fragments/wave); h in LDS (f16); c in registers.
// Gate pre-activations TDM'd into a 2-deep LDS buffer and summed in place.
// ---------------------------------------------------------------------------
__global__ __launch_bounds__(256) void k_rec(
    const _Float16* __restrict__ gpre,  // [M2,512] f16: x@w_ih^T + biases
    const _Float16* __restrict__ whh,   // [512,128] f16 (preconverted)
    _Float16*       __restrict__ hseq)  // [M2,128] f16 per-step h output
{
  __shared__ __align__(16) _Float16 h_lds[16][H_];       // 4 KB
  __shared__ __align__(16) _Float16 gbuf[2][16][G_];     // 32 KB double buffer
  const int b0   = blockIdx.x * 16;
  const int tid  = threadIdx.x;
  const int lane = tid & 31;
  const int wv   = tid >> 5;                     // wave 0..7 -> n-tiles wv*4..+3
  // persistent recurrent-weight fragments: 4 n-tiles x 4 K-chunks = 128 VGPRs
  v16h bfrag[4][4];
#pragma unroll
  for (int i = 0; i < 4; ++i)
#pragma unroll
    for (int ks = 0; ks < 4; ++ks)
      bfrag[i][ks] = load_b_frag(whh + (size_t)((wv * 4 + i) * 16 + (lane & 15)) * H_ + ks * 32, lane);
  for (int idx = tid; idx < 16 * H_; idx += 256) ((_Float16*)h_lds)[idx] = (_Float16)0.f;
  float c[8];
#pragma unroll
  for (int j = 0; j < 8; ++j) c[j] = 0.f;
  const int bl  = tid >> 4;                      // elementwise: batch-local row
  const int hu0 = tid & 15;                      // h-unit base (stride 16)
  // prologue: stage t=0
  stage_gpre(&gbuf[0][0][0], gpre + (size_t)b0 * T2_ * G_, tid);
  stage_wait(tid);
  __syncthreads();

  for (int t = 0; t < T_; ++t) {
    const int cur = t & 1;
    if (t + 1 < T_)  // prefetch next step's tile while we compute this one
      stage_gpre(&gbuf[cur ^ 1][0][0], gpre + ((size_t)b0 * T2_ + (t + 1)) * G_, tid);
    // --- gates(in place) = h @ w_hh^T (WMMA) + staged gpre ---
    v16h afrag[4];
    {
      const int hi = ((lane >> 4) & 1) << 3;
      const _Float16* hr = &h_lds[lane & 15][0];
#pragma unroll
      for (int ks = 0; ks < 4; ++ks) {
        const v8h* q = (const v8h*)(hr + ks * 32 + hi);
        afrag[ks] = cat16(q[0], q[2]);
      }
    }
#pragma unroll
    for (int i = 0; i < 4; ++i) {
      v8f acc = {};
#pragma unroll
      for (int ks = 0; ks < 4; ++ks)
        acc = __builtin_amdgcn_wmma_f32_16x16x32_f16(false, afrag[ks], false, bfrag[i][ks],
                                                     (short)0, acc, false, false);
      const int n = (wv * 4 + i) * 16 + (lane & 15);
#pragma unroll
      for (int r = 0; r < 8; ++r) {
        const int m = r + ((lane >> 4) << 3);
        gbuf[cur][m][n] = (_Float16)(acc[r] + (float)gbuf[cur][m][n]);
      }
    }
    __syncthreads();
    // --- elementwise cell update (i,f,g,o gate order) ---
#pragma unroll
    for (int j = 0; j < 8; ++j) {
      const int hu = hu0 + j * 16;
      float ig = (float)gbuf[cur][bl][hu];
      float fg = (float)gbuf[cur][bl][H_ + hu];
      float gg = (float)gbuf[cur][bl][2 * H_ + hu];
      float og = (float)gbuf[cur][bl][3 * H_ + hu];
      ig = 1.f / (1.f + __expf(-ig));
      fg = 1.f / (1.f + __expf(-fg));
      gg = tanhf(gg);
      og = 1.f / (1.f + __expf(-og));
      const float cc = fg * c[j] + ig * gg;
      c[j] = cc;
      const float hv = og * tanhf(cc);
      h_lds[bl][hu] = (_Float16)hv;
      hseq[((size_t)(b0 + bl) * T2_ + t) * H_ + hu] = (_Float16)hv;
    }
    stage_wait(tid);                 // next buffer is resident before reuse
    __syncthreads();
  }
}

} // namespace

extern "C" void kernel_launch(void* const* d_in, const int* in_sizes, int n_in,
                              void* d_out, int out_size, void* d_ws, size_t ws_size,
                              hipStream_t stream) {
  (void)in_sizes; (void)n_in; (void)out_size; (void)ws_size;
  const float* inputs = (const float*)d_in[0];
  const float* proj_w = (const float*)d_in[4];
  const float* proj_b = (const float*)d_in[5];
  const float* w_ih0  = (const float*)d_in[6];
  const float* w_hh0  = (const float*)d_in[7];
  const float* b_ih0  = (const float*)d_in[8];
  const float* b_hh0  = (const float*)d_in[9];
  const float* w_ih_r = (const float*)d_in[10];
  const float* w_hh_r = (const float*)d_in[11];
  const float* b_ih_r = (const float*)d_in[12];
  const float* b_hh_r = (const float*)d_in[13];
  const float* out_w  = (const float*)d_in[14];
  const float* out_b  = (const float*)d_in[15];
  float* out = (float*)d_out;

  char* ws = (char*)d_ws;
  _Float16* g  = (_Float16*)ws;                                        // [M2,512] f16
  _Float16* h  = (_Float16*)(ws + (size_t)M2_ * G_ * 2);               // [M2,128] f16
  _Float16* x0 = (_Float16*)(ws + (size_t)M2_ * (G_ + H_) * 2);        // [M2,320] f16
  // f16 weight staging area (1.66 MB) after the activation buffers
  _Float16* wf = (_Float16*)(ws + (size_t)M2_ * (G_ + H_ + DP_) * 2);
  _Float16* pw16   = wf;                           // 320*640
  _Float16* wih016 = pw16   + (size_t)DP_ * DIN_;  // 512*320
  _Float16* whh016 = wih016 + (size_t)G_ * DP_;    // 512*128
  _Float16* wihr16 = whh016 + (size_t)G_ * H_;     // 3*512*128
  _Float16* whhr16 = wihr16 + (size_t)3 * G_ * H_; // 3*512*128
  _Float16* oww16  = whhr16 + (size_t)3 * G_ * H_; // 48*128

  // 0. one-shot weight f32->f16 conversion
  k_cvt<<<dim3((DP_ * DIN_ + 255) / 256), dim3(256), 0, stream>>>(proj_w, pw16, DP_ * DIN_);
  k_cvt<<<dim3((G_ * DP_ + 255) / 256), dim3(256), 0, stream>>>(w_ih0, wih016, G_ * DP_);
  k_cvt<<<dim3((G_ * H_ + 255) / 256), dim3(256), 0, stream>>>(w_hh0, whh016, G_ * H_);
  k_cvt<<<dim3((3 * G_ * H_ + 255) / 256), dim3(256), 0, stream>>>(w_ih_r, wihr16, 3 * G_ * H_);
  k_cvt<<<dim3((3 * G_ * H_ + 255) / 256), dim3(256), 0, stream>>>(w_hh_r, whhr16, 3 * G_ * H_);
  k_cvt<<<dim3((LBL_ * H_ + 255) / 256), dim3(256), 0, stream>>>(out_w, oww16, LBL_ * H_);

  const int mt = (int)(M2_ / 16);  // 8128 M-tiles

  // 1. fused unfold + projection -> x0 (f16)
  k_proj<<<dim3(mt * (DP_ / 32) / 8), dim3(256), 0, stream>>>(inputs, pw16, proj_b, x0);
  // 2. layer-0 input precompute -> g (f16), NBLK=4
  k_gemm<_Float16, _Float16, 4><<<dim3(mt * (G_ / 64) / 8), dim3(256), 0, stream>>>(
      x0, DP_, wih016, DP_, b_ih0, b_hh0, g, G_, mt, G_ / 64, DP_);
  // 3. layer-0 recurrence -> h
  k_rec<<<dim3(4), dim3(256), 0, stream>>>(g, whh016, h);
  // 4. layers 1..3
  for (int l = 0; l < 3; ++l) {
    k_gemm<_Float16, _Float16, 4><<<dim3(mt * (G_ / 64) / 8), dim3(256), 0, stream>>>(
        h, H_, wihr16 + (size_t)l * G_ * H_, H_,
        b_ih_r + (size_t)l * G_, b_hh_r + (size_t)l * G_, g, G_, mt, G_ / 64, H_);
    k_rec<<<dim3(4), dim3(256), 0, stream>>>(g, whhr16 + (size_t)l * G_ * H_, h);
  }
  // 5. head: out[64,48] = h[:,T-1,:] @ out_w^T + out_b (NBLK=3 covers N=48)
  k_gemm<_Float16, float, 3><<<dim3(1), dim3(256), 0, stream>>>(
      h + (size_t)(T_ - 1) * H_, (long)T2_ * H_, oww16, H_,
      out_b, nullptr, out, LBL_, B_ / 16, 1, H_);
}
